// SwinTransformerBlock_50895362457911
// MI455X (gfx1250) — compile-verified, matrix operand order checked by reading, not run on hardware
//
#include <hip/hip_runtime.h>
#include <math.h>

// ---------------- static problem sizes ----------------
#define BATCH   64
#define HRES    56
#define CDIM    192
#define HEADS   6
#define HD      32
#define WSZ     7
#define SSZ     3
#define NWINROW 8                       // 56/7
#define NWIN    (BATCH * NWINROW * NWINROW)   // 4096
#define NTOK_W  49                      // tokens per window
#define TOK     (NWIN * NTOK_W)         // 200704 windowed tokens (== B*L)
#define LSEQ    (HRES * HRES)           // 3136
#define HID     768
#define QKVN    (3 * CDIM)              // 576

typedef __attribute__((ext_vector_type(16))) _Float16 v16h;
typedef __attribute__((ext_vector_type(8)))  float    v8f;

// ---------------- WMMA helpers ----------------
__device__ inline v8f wmma16(v16h a, v16h b, v8f c) {
  // D = A(16x32 f16) x B(32x16 f16) + C(16x16 f32)
  return __builtin_amdgcn_wmma_f32_16x16x32_f16(
      /*neg_a=*/false, a, /*neg_b=*/false, b,
      /*c_mod=*/(short)0, c, /*reuse_a=*/false, /*reuse_b=*/false);
}

// Load one 16x32 f16 fragment. Row index = lane%16, K halves:
//   halves 0..7  -> K in [8g, 8g+8)
//   halves 8..15 -> K in [16+8g, 16+8g+8)   (g = lane/16)
// Works for A tiles (row-major M x K) and for B tiles stored N-major
// with K contiguous (i.e. weight matrices W[out][in]).
__device__ inline v16h frag_ld(const _Float16* p, int stride) {
  int lane = threadIdx.x & 31;
  const _Float16* q = p + (long)(lane & 15) * stride + (lane >> 4) * 8;
  v16h r;
  ((uint4*)&r)[0] = *(const uint4*)(q);
  ((uint4*)&r)[1] = *(const uint4*)(q + 16);
  return r;
}

// CDNA5 async bulk staging: global -> LDS without VGPR round-trip.
// Tracked by ASYNCcnt; wave waits, then workgroup barrier makes LDS visible.
__device__ inline void stage_async_tile(const _Float16* __restrict__ g,
                                        _Float16* lds, int n_chunks16) {
  uint32_t lbase = (uint32_t)(uintptr_t)lds;   // low 32 bits of generic ptr = LDS byte offset
  for (int i = threadIdx.x; i < n_chunks16; i += blockDim.x) {
    const _Float16* ga = g + i * 8;            // 8 halfs = 16 bytes per chunk
    uint32_t la = lbase + i * 16;
    asm volatile("global_load_async_to_lds_b128 %0, %1, off"
                 :: "v"(la), "v"(ga) : "memory");
  }
  asm volatile("s_wait_asynccnt 0" ::: "memory");
  __syncthreads();
}

__device__ inline int reg3(int a) {   // Swin shift-mask region id along one axis
  return a < (HRES - WSZ) ? 0 : (a < (HRES - SSZ) ? 1 : 2);
}

// ---------------- 1) weights f32 -> f16 ----------------
__global__ void cvt_weights_kernel(const float* qkvw, const float* projw,
                                   const float* w1, const float* w2,
                                   _Float16* oq, _Float16* op,
                                   _Float16* o1, _Float16* o2) {
  int i = blockIdx.x * 256 + threadIdx.x;
  if (i < QKVN * CDIM) oq[i] = (_Float16)qkvw[i];
  if (i < CDIM * CDIM) op[i] = (_Float16)projw[i];
  if (i < HID * CDIM)  { o1[i] = (_Float16)w1[i]; o2[i] = (_Float16)w2[i]; }
}

// ---------------- 2) LN1 + roll(-3,-3) + window partition -> f16 ----------------
__global__ void ln1_window_kernel(const float* __restrict__ x,
                                  const float* __restrict__ gam,
                                  const float* __restrict__ bet,
                                  _Float16* __restrict__ out) {
  int wid  = threadIdx.x >> 5;
  int lane = threadIdx.x & 31;
  long tw = (long)blockIdx.x * 8 + wid;          // windowed token id
  int win = (int)(tw / NTOK_W), p = (int)(tw % NTOK_W);
  int bb = win >> 6, wr = win & 63, wy = wr >> 3, wx = wr & 7;
  int iy = p / WSZ, ix = p % WSZ;
  int h = (wy * WSZ + iy + SSZ) % HRES;          // roll(-s): shifted <- src
  int w = (wx * WSZ + ix + SSZ) % HRES;
  const float* xr = x + ((long)bb * LSEQ + h * HRES + w) * CDIM;
  float v[6], s = 0.f, s2 = 0.f;
#pragma unroll
  for (int j = 0; j < 6; j++) { float t = xr[lane + 32 * j]; v[j] = t; s += t; s2 += t * t; }
#pragma unroll
  for (int d = 16; d >= 1; d >>= 1) { s += __shfl_xor(s, d, 32); s2 += __shfl_xor(s2, d, 32); }
  float mu = s * (1.0f / CDIM);
  float rs = rsqrtf(s2 * (1.0f / CDIM) - mu * mu + 1e-5f);
  _Float16* o = out + tw * CDIM;
#pragma unroll
  for (int j = 0; j < 6; j++) {
    int c = lane + 32 * j;
    o[c] = (_Float16)((v[j] - mu) * rs * gam[c] + bet[c]);
  }
}

// ---------------- 3) QKV GEMM: (TOK x 192) @ (192 x 576) + bias -> f16 ----------------
__global__ void gemm_qkv_kernel(const _Float16* __restrict__ A,
                                const _Float16* __restrict__ W,   // 576 x 192 (N-major,K-contig)
                                const float* __restrict__ bias,
                                _Float16* __restrict__ out) {
  __shared__ _Float16 a_lds[64 * CDIM];
  long row0 = (long)blockIdx.x * 64;
  int  col0 = blockIdx.y * 64;
  stage_async_tile(A + row0 * CDIM, a_lds, 64 * CDIM / 8);
  int wv = threadIdx.x >> 5, lane = threadIdx.x & 31;
  const _Float16* abase = a_lds + wv * 16 * CDIM;
  v16h afr[6];                          // wave's A stripe, loaded once, reused 4x
#pragma unroll
  for (int k = 0; k < 6; k++) afr[k] = frag_ld(abase + k * 32, CDIM);
#pragma unroll
  for (int tn = 0; tn < 4; tn++) {
    int col = col0 + tn * 16 + (lane & 15);
    const _Float16* wb = W + (long)(col0 + tn * 16) * CDIM;
    v8f acc = {};
#pragma unroll
    for (int k = 0; k < 6; k++)
      acc = wmma16(afr[k], frag_ld(wb + k * 32, CDIM), acc);
    float bv = bias[col];
    long orow = row0 + wv * 16 + 8 * (lane >> 4);
#pragma unroll
    for (int r = 0; r < 8; r++)
      out[(orow + r) * QKVN + col] = (_Float16)(acc[r] + bv);
  }
}

// ---------------- 4) windowed attention: one wave per (window, head) ----------------
__global__ void attn_kernel(const _Float16* __restrict__ qkv,
                            const float* __restrict__ bias_table,  // 169 x 6
                            _Float16* __restrict__ outA) {
  __shared__ _Float16 q_s[64 * HD];
  __shared__ _Float16 k_s[64 * HD];
  __shared__ _Float16 vT_s[HD * 64];
  __shared__ _Float16 a_s[64 * 64];
  int bid = blockIdx.x;
  int win = bid / HEADS, head = bid % HEADS;
  int lane = threadIdx.x & 31;
  int wr = win & 63, wy = wr >> 3, wx = wr & 7;
  long base = (long)win * NTOK_W;

  // stage Q,K (zero-pad rows 49..63)
  for (int idx = lane; idx < 64 * 4; idx += 32) {
    int row = idx >> 2, ch = idx & 3;
    uint4 z = {0u, 0u, 0u, 0u}, qv = z, kv = z;
    if (row < NTOK_W) {
      const _Float16* s = qkv + (base + row) * QKVN + head * HD + ch * 8;
      qv = *(const uint4*)(s);
      kv = *(const uint4*)(s + CDIM);
    }
    *(uint4*)(q_s + row * HD + ch * 8) = qv;
    *(uint4*)(k_s + row * HD + ch * 8) = kv;
  }
  // stage V transposed: vT[d][key], d = lane
  for (int key = 0; key < 64; key++) {
    _Float16 vv = (_Float16)0.0f;
    if (key < NTOK_W) vv = qkv[(base + key) * QKVN + 2 * CDIM + head * HD + lane];
    vT_s[lane * 64 + key] = vv;
  }
  __syncthreads();

  const float scale = 0.17677669529663687f;   // 32^-0.5
  int g = lane >> 4, nn = lane & 15;

  // K fragments reused by every row stripe: load once
  v16h kf[4];
#pragma unroll
  for (int tn = 0; tn < 4; tn++) kf[tn] = frag_ld(k_s + tn * 16 * HD, HD);

  for (int tm = 0; tm < 4; tm++) {
    v16h aF = frag_ld(q_s + tm * 16 * HD, HD);   // K = hd = 32: single WMMA step
    v8f sc[4];
#pragma unroll
    for (int tn = 0; tn < 4; tn++) {
      v8f z = {};
      sc[tn] = wmma16(aF, kf[tn], z);
    }
    // scale + relative-position bias + shift mask (analytic)
#pragma unroll
    for (int tn = 0; tn < 4; tn++) {
      int qq = tn * 16 + nn;
#pragma unroll
      for (int r = 0; r < 8; r++) {
        int pp = tm * 16 + r + 8 * g;
        float s;
        if (pp < NTOK_W && qq < NTOK_W) {
          int iyp = pp / WSZ, ixp = pp % WSZ, iyq = qq / WSZ, ixq = qq % WSZ;
          int ridx = (iyp - iyq + WSZ - 1) * (2 * WSZ - 1) + (ixp - ixq + WSZ - 1);
          float bias = bias_table[ridx * HEADS + head];
          int cp = reg3(wy * WSZ + iyp) * 3 + reg3(wx * WSZ + ixp);
          int cq = reg3(wy * WSZ + iyq) * 3 + reg3(wx * WSZ + ixq);
          s = sc[tn][r] * scale + bias + ((cp != cq) ? -100.0f : 0.0f);
        } else {
          s = -1e30f;
        }
        sc[tn][r] = s;
      }
    }
    // row softmax: 4 tiles in-register + shfl over the 16-lane half
#pragma unroll
    for (int r = 0; r < 8; r++) {
      float m = fmaxf(fmaxf(sc[0][r], sc[1][r]), fmaxf(sc[2][r], sc[3][r]));
#pragma unroll
      for (int d = 8; d >= 1; d >>= 1) m = fmaxf(m, __shfl_xor(m, d, 32));
      float e0 = expf(sc[0][r] - m), e1 = expf(sc[1][r] - m);
      float e2 = expf(sc[2][r] - m), e3 = expf(sc[3][r] - m);
      float sum = e0 + e1 + e2 + e3;
#pragma unroll
      for (int d = 8; d >= 1; d >>= 1) sum += __shfl_xor(sum, d, 32);
      float inv = 1.0f / sum;
      sc[0][r] = e0 * inv; sc[1][r] = e1 * inv; sc[2][r] = e2 * inv; sc[3][r] = e3 * inv;
    }
    // spill attn stripe to LDS (row-major, K-contig for the next A-frag)
#pragma unroll
    for (int tn = 0; tn < 4; tn++) {
      int qq = tn * 16 + nn;
#pragma unroll
      for (int r = 0; r < 8; r++) {
        int pp = tm * 16 + r + 8 * g;
        a_s[pp * 64 + qq] = (_Float16)sc[tn][r];
      }
    }
  }
  __syncthreads();

  // out = attn(64x64) @ V(64x32): K = 64 -> 2 WMMA steps, N = 32 -> 2 col tiles
  v16h vf[2][2];                        // V fragments reused by every row stripe
#pragma unroll
  for (int tn = 0; tn < 2; tn++)
#pragma unroll
    for (int kk = 0; kk < 2; kk++)
      vf[tn][kk] = frag_ld(vT_s + tn * 16 * 64 + kk * 32, 64);

  for (int tm = 0; tm < 4; tm++) {
    v8f o0 = {}, o1 = {};
#pragma unroll
    for (int kk = 0; kk < 2; kk++) {
      v16h aF = frag_ld(a_s + tm * 16 * 64 + kk * 32, 64);
      o0 = wmma16(aF, vf[0][kk], o0);
      o1 = wmma16(aF, vf[1][kk], o1);
    }
#pragma unroll
    for (int r = 0; r < 8; r++) {
      int pp = tm * 16 + r + 8 * g;
      if (pp < NTOK_W) {
        long ob = (base + pp) * CDIM + head * HD;
        outA[ob + nn]      = (_Float16)o0[r];
        outA[ob + 16 + nn] = (_Float16)o1[r];
      }
    }
  }
}

// ---------------- 5) proj GEMM + window reverse + roll(+3) + residual ----------------
__global__ void gemm_proj_kernel(const _Float16* __restrict__ A,     // TOK x 192 (windowed)
                                 const _Float16* __restrict__ W,     // 192 x 192
                                 const float* __restrict__ bias,
                                 const float* __restrict__ x,        // shortcut
                                 float* __restrict__ outp) {
  __shared__ _Float16 a_lds[64 * CDIM];
  long row0 = (long)blockIdx.x * 64;
  int  col0 = blockIdx.y * 64;
  stage_async_tile(A + row0 * CDIM, a_lds, 64 * CDIM / 8);
  int wv = threadIdx.x >> 5, lane = threadIdx.x & 31;
  const _Float16* abase = a_lds + wv * 16 * CDIM;
  v16h afr[6];
#pragma unroll
  for (int k = 0; k < 6; k++) afr[k] = frag_ld(abase + k * 32, CDIM);
#pragma unroll
  for (int tn = 0; tn < 4; tn++) {
    int col = col0 + tn * 16 + (lane & 15);
    const _Float16* wb = W + (long)(col0 + tn * 16) * CDIM;
    v8f acc = {};
#pragma unroll
    for (int k = 0; k < 6; k++)
      acc = wmma16(afr[k], frag_ld(wb + k * 32, CDIM), acc);
    float bv = bias[col];
    long trow = row0 + wv * 16 + 8 * (lane >> 4);
#pragma unroll
    for (int r = 0; r < 8; r++) {
      long tw = trow + r;
      int win = (int)(tw / NTOK_W), p = (int)(tw % NTOK_W);
      int bb = win >> 6, wr = win & 63, wy = wr >> 3, wx = wr & 7;
      int iy = p / WSZ, ix = p % WSZ;
      int h = (wy * WSZ + iy + SSZ) % HRES;   // reverse roll lands back at source
      int w = (wx * WSZ + ix + SSZ) % HRES;
      long o = ((long)bb * LSEQ + h * HRES + w) * CDIM + col;
      outp[o] = x[o] + acc[r] + bv;           // x1 = shortcut + attn_out
    }
  }
}

// ---------------- 6) LN2 (token order) -> f16 ----------------
__global__ void ln2_kernel(const float* __restrict__ x1,
                           const float* __restrict__ gam,
                           const float* __restrict__ bet,
                           _Float16* __restrict__ out) {
  int wid = threadIdx.x >> 5, lane = threadIdx.x & 31;
  long t = (long)blockIdx.x * 8 + wid;
  const float* xr = x1 + t * CDIM;
  float v[6], s = 0.f, s2 = 0.f;
#pragma unroll
  for (int j = 0; j < 6; j++) { float u = xr[lane + 32 * j]; v[j] = u; s += u; s2 += u * u; }
#pragma unroll
  for (int d = 16; d >= 1; d >>= 1) { s += __shfl_xor(s, d, 32); s2 += __shfl_xor(s2, d, 32); }
  float mu = s * (1.0f / CDIM);
  float rs = rsqrtf(s2 * (1.0f / CDIM) - mu * mu + 1e-5f);
  _Float16* o = out + t * CDIM;
#pragma unroll
  for (int j = 0; j < 6; j++) {
    int c = lane + 32 * j;
    o[c] = (_Float16)((v[j] - mu) * rs * gam[c] + bet[c]);
  }
}

// ---------------- 7) fused MLP: fc1 + GELU (LDS-resident hidden) + fc2 + residual ----------------
__global__ void mlp_kernel(const _Float16* __restrict__ A,     // TOK x 192 (LN2 out)
                           const _Float16* __restrict__ W1,    // 768 x 192
                           const float* __restrict__ b1,
                           const _Float16* __restrict__ W2,    // 192 x 768
                           const float* __restrict__ b2,
                           float* __restrict__ outp) {         // x1 in, x1 + mlp out
  extern __shared__ _Float16 smem[];
  _Float16* a_lds = smem;                 // 64 x 192  (24 KB)
  _Float16* hid   = smem + 64 * CDIM;     // 64 x 768  (96 KB)
  long row0 = (long)blockIdx.x * 64;
  stage_async_tile(A + row0 * CDIM, a_lds, 64 * CDIM / 8);
  int wv = threadIdx.x >> 5, lane = threadIdx.x & 31;
  const _Float16* abase = a_lds + wv * 16 * CDIM;
  int rbase = wv * 16 + 8 * (lane >> 4);

  // fc1: 64 x 768 hidden tile, GELU, kept in LDS.
  // A-frags loaded once per wave, reused across all 48 column tiles.
  v16h afr[6];
#pragma unroll
  for (int k = 0; k < 6; k++) afr[k] = frag_ld(abase + k * 32, CDIM);
  for (int tn = 0; tn < HID / 16; tn++) {
    int col = tn * 16 + (lane & 15);
    const _Float16* wb = W1 + (long)tn * 16 * CDIM;
    v8f acc = {};
#pragma unroll
    for (int k = 0; k < 6; k++)
      acc = wmma16(afr[k], frag_ld(wb + k * 32, CDIM), acc);
    float bv = b1[col];
#pragma unroll
    for (int r = 0; r < 8; r++) {
      float t = acc[r] + bv;
      float gl = 0.5f * t * (1.0f + erff(t * 0.70710678118654752f));   // exact GELU
      hid[(rbase + r) * HID + col] = (_Float16)gl;
    }
  }
  __syncthreads();

  // fc2: 64 x 192, K = 768. Loop k outermost with 12 live accumulators so each
  // hidden A-frag is read from LDS exactly once.
  const _Float16* hbase = hid + wv * 16 * HID;
  v8f acc[CDIM / 16];
#pragma unroll
  for (int tn = 0; tn < CDIM / 16; tn++) acc[tn] = (v8f){};
  for (int k = 0; k < 24; k++) {
    v16h a = frag_ld(hbase + k * 32, HID);
#pragma unroll
    for (int tn = 0; tn < CDIM / 16; tn++)
      acc[tn] = wmma16(a, frag_ld(W2 + (long)tn * 16 * HID + k * 32, HID), acc[tn]);
  }
#pragma unroll
  for (int tn = 0; tn < CDIM / 16; tn++) {
    int col = tn * 16 + (lane & 15);
    float bv = b2[col];
    long orow = row0 + rbase;
#pragma unroll
    for (int r = 0; r < 8; r++) {
      long o = (orow + r) * CDIM + col;
      outp[o] = outp[o] + acc[tn][r] + bv;
    }
  }
}

// ---------------- launcher ----------------
extern "C" void kernel_launch(void* const* d_in, const int* in_sizes, int n_in,
                              void* d_out, int out_size, void* d_ws, size_t ws_size,
                              hipStream_t stream) {
  const float* x        = (const float*)d_in[0];
  const float* n1g      = (const float*)d_in[1];
  const float* n1b      = (const float*)d_in[2];
  const float* qkv_w    = (const float*)d_in[3];
  const float* qkv_b    = (const float*)d_in[4];
  const float* rel_bias = (const float*)d_in[5];
  const float* proj_w   = (const float*)d_in[6];
  const float* proj_b   = (const float*)d_in[7];
  const float* n2g      = (const float*)d_in[8];
  const float* n2b      = (const float*)d_in[9];
  const float* fc1_w    = (const float*)d_in[10];
  const float* fc1_b    = (const float*)d_in[11];
  const float* fc2_w    = (const float*)d_in[12];
  const float* fc2_b    = (const float*)d_in[13];
  float* out = (float*)d_out;

  // scratch layout (regions are re-used once their producer/consumer chain ends)
  char* ws = (char*)d_ws;
  _Float16* s_xw  = (_Float16*)ws;                              // TOK*192 f16 (LN1 windows; reused for attn out)
  _Float16* s_att = s_xw;
  _Float16* s_qkv = (_Float16*)(ws + (size_t)TOK * CDIM * 2);   // TOK*576 f16 (reused for LN2 out)
  _Float16* s_h2  = s_qkv;
  char* wsw = ws + (size_t)TOK * CDIM * 2 + (size_t)TOK * QKVN * 2;
  _Float16* wq16 = (_Float16*)wsw;
  _Float16* wp16 = wq16 + QKVN * CDIM;
  _Float16* w116 = wp16 + CDIM * CDIM;
  _Float16* w216 = w116 + HID * CDIM;

  cvt_weights_kernel<<<(HID * CDIM + 255) / 256, 256, 0, stream>>>(
      qkv_w, proj_w, fc1_w, fc2_w, wq16, wp16, w116, w216);

  ln1_window_kernel<<<TOK / 8, 256, 0, stream>>>(x, n1g, n1b, s_xw);

  gemm_qkv_kernel<<<dim3(TOK / 64, QKVN / 64), 128, 0, stream>>>(
      s_xw, wq16, qkv_b, s_qkv);

  attn_kernel<<<NWIN * HEADS, 32, 0, stream>>>(s_qkv, rel_bias, s_att);

  gemm_proj_kernel<<<dim3(TOK / 64, CDIM / 64), 128, 0, stream>>>(
      s_att, wp16, proj_b, x, out);

  ln2_kernel<<<TOK / 8, 256, 0, stream>>>(out, n2g, n2b, s_h2);

  size_t mlp_lds = (size_t)(64 * CDIM + 64 * HID) * sizeof(_Float16);  // 120 KB < 320 KB WGP LDS
  mlp_kernel<<<TOK / 64, 128, mlp_lds, stream>>>(s_h2, w116, fc1_b, w216, fc2_b, out);
}
